// CrossAttentionBlock_6640019439884
// MI455X (gfx1250) — matrix operand order repaired and checked by reading, not compiled, not run on hardware
//
#include <hip/hip_runtime.h>

typedef __attribute__((ext_vector_type(16))) _Float16 v16h;
typedef __attribute__((ext_vector_type(8)))  float    v8f;

#define WMMA_F16(a, b, c) \
  __builtin_amdgcn_wmma_f32_16x16x32_f16(false, (a), false, (b), (short)0, (c), false, false)

// ---------------------------------------------------------------------------
// Fragment loaders for gfx1250 wave32 WMMA (16x16x32 f16).
// A (16Mx32K, row-major source, K contiguous):
//   lane<16 : row M=lane,     halves = K[0..7] then K[16..23]
//   lane>=16: row M=lane-16,  halves = K[8..15] then K[24..31]
static __device__ __forceinline__ v16h load_frag_a(const _Float16* base, int ld) {
  const int lane = threadIdx.x & 31;
  const int row  = lane & 15;
  const int k0   = (lane & 16) ? 8 : 0;
  const _Float16* p = base + (size_t)row * ld;
  v16h a;
#pragma unroll
  for (int t = 0; t < 8; ++t) a[t] = p[k0 + t];
#pragma unroll
  for (int t = 0; t < 8; ++t) a[8 + t] = p[k0 + 16 + t];
  return a;
}

// B (32Kx16N, [K][N] source, N contiguous): lane L holds column N=L%16, K same order as A.
static __device__ __forceinline__ v16h load_frag_b(const _Float16* base, int ld) {
  const int lane = threadIdx.x & 31;
  const int k0   = (lane & 16) ? 8 : 0;
  const _Float16* p = base + (lane & 15);
  v16h b;
#pragma unroll
  for (int t = 0; t < 8; ++t) b[t] = p[(size_t)(k0 + t) * ld];
#pragma unroll
  for (int t = 0; t < 8; ++t) b[8 + t] = p[(size_t)(k0 + 16 + t) * ld];
  return b;
}
// C/D frag: lane<16 -> N=lane, VGPR r -> M=r ; lane>=16 -> N=lane-16, M=8+r.
// ---------------------------------------------------------------------------

__global__ void f2h_kernel(const float* __restrict__ s, _Float16* __restrict__ d, int n) {
  int i = blockIdx.x * blockDim.x + threadIdx.x;
  if (i < n) d[i] = (_Float16)s[i];
}

// GroupNorm: one block per (batch, group); 16 channels x 4096 px = 65536 elems.
__global__ __launch_bounds__(256) void gn_kernel(const float* __restrict__ x,
                                                 const float* __restrict__ gw,
                                                 const float* __restrict__ gb,
                                                 _Float16* __restrict__ xn) {
  const int N = 4096, CPG = 16;
  const int bg = blockIdx.x, batch = bg >> 5, g = bg & 31;
  const float* xp = x + ((size_t)batch * 512 + g * CPG) * N;
  _Float16*    xo = xn + ((size_t)batch * 512 + g * CPG) * N;
  const int cnt = CPG * N;  // 65536
  float s = 0.f, s2 = 0.f;
  for (int i = threadIdx.x; i < cnt; i += 256) { float v = xp[i]; s += v; s2 += v * v; }
#pragma unroll
  for (int m = 16; m >= 1; m >>= 1) { s += __shfl_xor(s, m, 32); s2 += __shfl_xor(s2, m, 32); }
  __shared__ float rs[8], rs2[8];
  const int wid = threadIdx.x >> 5;
  if ((threadIdx.x & 31) == 0) { rs[wid] = s; rs2[wid] = s2; }
  __syncthreads();
  float ts = 0.f, ts2 = 0.f;
#pragma unroll
  for (int w = 0; w < 8; ++w) { ts += rs[w]; ts2 += rs2[w]; }
  const float invn = 1.0f / 65536.0f;
  const float mu  = ts * invn;
  const float var = ts2 * invn - mu * mu;
  const float inv = rsqrtf(var + 1e-5f);
  for (int i = threadIdx.x; i < cnt; i += 256) {
    const int cg = g * CPG + (i >> 12);
    xo[i] = (_Float16)((xp[i] - mu) * inv * gw[cg] + gb[cg]);
  }
}

// 1x1 conv as GEMM, register-blocked: one wave -> 32(M) x 64(N) output tile.
// Per k-chunk: 2 A-frags + 4 B-frags -> 8 WMMA.
// mode 0: store [b][c][p]; mode 1: store [b][h][p][d] (per-head transposed).
__global__ __launch_bounds__(32) void proj_kernel(const _Float16* __restrict__ W,
                                                  const float* __restrict__ bias,
                                                  const _Float16* __restrict__ xn,
                                                  _Float16* __restrict__ out, int mode) {
  const int N = 4096, C = 512;
  const int ct = blockIdx.x;            // 128 col blocks over B*N (64 per batch)
  const int ot = blockIdx.y;            // 16 row blocks
  const int batch = ct >> 6;
  const int p0 = (ct & 63) * 64;
  const int o0 = ot * 32;
  const _Float16* xb = xn + (size_t)batch * C * N;

  v8f acc[2][4];
  {
    v8f z = {};
#pragma unroll
    for (int i = 0; i < 2; ++i)
#pragma unroll
      for (int j = 0; j < 4; ++j) acc[i][j] = z;
  }
  for (int kc = 0; kc < C; kc += 32) {
    v16h a0 = load_frag_a(W + (size_t)(o0 + 0)  * C + kc, C);
    v16h a1 = load_frag_a(W + (size_t)(o0 + 16) * C + kc, C);
#pragma unroll
    for (int j = 0; j < 4; ++j) {
      v16h b = load_frag_b(xb + (size_t)kc * N + p0 + 16 * j, N);
      acc[0][j] = WMMA_F16(a0, b, acc[0][j]);
      acc[1][j] = WMMA_F16(a1, b, acc[1][j]);
    }
  }
  const int lane = threadIdx.x, n = lane & 15, mb = (lane & 16) ? 8 : 0;
#pragma unroll
  for (int i = 0; i < 2; ++i) {
#pragma unroll
    for (int j = 0; j < 4; ++j) {
#pragma unroll
      for (int r = 0; r < 8; ++r) {
        const int o = o0 + 16 * i + mb + r;
        const int p = p0 + 16 * j + n;
        const float val = acc[i][j][r] + bias[o];
        if (mode == 0) {
          out[((size_t)batch * C + o) * N + p] = (_Float16)val;
        } else {
          const int h = o >> 6, d = o & 63;
          out[(((size_t)batch * 8 + h) * N + p) * 64 + d] = (_Float16)val;
        }
      }
    }
  }
}

// Flash attention: one wave per (b, h, 32-row i-block = two 16-row i-tiles).
// Every K/V B-fragment is reused for both i-tiles (2 WMMA per fragment load).
__global__ __launch_bounds__(32) void attn_kernel(const _Float16* __restrict__ q_t,
                                                  const _Float16* __restrict__ k_buf,
                                                  const _Float16* __restrict__ v_t,
                                                  _Float16* __restrict__ o_buf) {
  const int N = 4096;
  const int ib = blockIdx.x;            // 128 i-blocks of 32 rows
  const int bh = blockIdx.y;            // b*8+h
  const int batch = bh >> 3, h = bh & 7;
  const int i0 = ib * 32;
  const int lane = threadIdx.x, n = lane & 15, mb = (lane & 16) ? 8 : 0;

  const _Float16* kp = k_buf + ((size_t)batch * 512 + h * 64) * N; // [d][j], ld N
  const _Float16* vp = v_t + (size_t)bh * N * 64;                  // [j][d], ld 64

  v16h aq[2][2];
#pragma unroll
  for (int it = 0; it < 2; ++it) {
    const _Float16* qp = q_t + ((size_t)bh * N + i0 + it * 16) * 64; // [i][d], ld 64
    aq[it][0] = load_frag_a(qp, 64);       // d = 0..31
    aq[it][1] = load_frag_a(qp + 32, 64);  // d = 32..63
  }

  __shared__ _Float16 pl[2][16 * 32];      // P tiles, row-major [i][j_local]

  float m_run[2][8], l_run[2][8];
  v8f oacc[2][4];
  {
    v8f z = {};
#pragma unroll
    for (int it = 0; it < 2; ++it) {
#pragma unroll
      for (int t = 0; t < 4; ++t) oacc[it][t] = z;
#pragma unroll
      for (int r = 0; r < 8; ++r) { m_run[it][r] = -3.0e38f; l_run[it][r] = 0.f; }
    }
  }
  const float scale = 0.125f;  // 64^-0.5

  for (int j0 = 0; j0 < N; j0 += 32) {
    v8f s0[2], s1[2];
    {
      v8f z = {};
#pragma unroll
      for (int it = 0; it < 2; ++it) { s0[it] = z; s1[it] = z; }
    }
    {  // S = Q^T K for both i-tiles; each K fragment reused twice.
      v16h bk = load_frag_b(kp + j0, N);                      // d 0..31,  j0..j0+15
      s0[0] = WMMA_F16(aq[0][0], bk, s0[0]);
      s0[1] = WMMA_F16(aq[1][0], bk, s0[1]);
      bk = load_frag_b(kp + (size_t)32 * N + j0, N);          // d 32..63, j0..j0+15
      s0[0] = WMMA_F16(aq[0][1], bk, s0[0]);
      s0[1] = WMMA_F16(aq[1][1], bk, s0[1]);
      bk = load_frag_b(kp + j0 + 16, N);                      // d 0..31,  j0+16..31
      s1[0] = WMMA_F16(aq[0][0], bk, s1[0]);
      s1[1] = WMMA_F16(aq[1][0], bk, s1[1]);
      bk = load_frag_b(kp + (size_t)32 * N + j0 + 16, N);     // d 32..63, j0+16..31
      s1[0] = WMMA_F16(aq[0][1], bk, s1[0]);
      s1[1] = WMMA_F16(aq[1][1], bk, s1[1]);
    }
    // Online softmax per i-tile; P written to LDS in A-frag source layout.
#pragma unroll
    for (int it = 0; it < 2; ++it) {
      float alpha[8];
#pragma unroll
      for (int r = 0; r < 8; ++r) {
        const float a0 = s0[it][r] * scale, a1 = s1[it][r] * scale;
        float mc = fmaxf(a0, a1);
#pragma unroll
        for (int msk = 1; msk <= 8; msk <<= 1) mc = fmaxf(mc, __shfl_xor(mc, msk, 32));
        const float mnew = fmaxf(m_run[it][r], mc);
        const float al = __expf(m_run[it][r] - mnew);
        const float p0 = __expf(a0 - mnew);
        const float p1 = __expf(a1 - mnew);
        float ps = p0 + p1;
#pragma unroll
        for (int msk = 1; msk <= 8; msk <<= 1) ps += __shfl_xor(ps, msk, 32);
        l_run[it][r] = al * l_run[it][r] + ps;
        m_run[it][r] = mnew;
        alpha[r] = al;
        pl[it][(mb + r) * 32 + n]      = (_Float16)p0;
        pl[it][(mb + r) * 32 + n + 16] = (_Float16)p1;
      }
#pragma unroll
      for (int t = 0; t < 4; ++t)
#pragma unroll
        for (int r = 0; r < 8; ++r) oacc[it][t][r] *= alpha[r];
    }
    __syncthreads();
    v16h ap0 = load_frag_a(pl[0], 32);
    v16h ap1 = load_frag_a(pl[1], 32);
#pragma unroll
    for (int t = 0; t < 4; ++t) {  // each V fragment reused twice
      v16h bv = load_frag_b(vp + (size_t)j0 * 64 + t * 16, 64);
      oacc[0][t] = WMMA_F16(ap0, bv, oacc[0][t]);
      oacc[1][t] = WMMA_F16(ap1, bv, oacc[1][t]);
    }
    __syncthreads();
  }
  // Normalize and scatter to [b][c][i] so the out-projection B-frag is contiguous.
#pragma unroll
  for (int it = 0; it < 2; ++it) {
#pragma unroll
    for (int t = 0; t < 4; ++t) {
#pragma unroll
      for (int r = 0; r < 8; ++r) {
        const float val = oacc[it][t][r] / l_run[it][r];
        const int i = i0 + it * 16 + mb + r;
        const int c = h * 64 + t * 16 + n;
        o_buf[((size_t)batch * 512 + c) * N + i] = (_Float16)val;
      }
    }
  }
}

// Output projection + bias + residual, f32 result. Same 32x64 register blocking.
__global__ __launch_bounds__(32) void outproj_kernel(const _Float16* __restrict__ W,
                                                     const float* __restrict__ bias,
                                                     const _Float16* __restrict__ ob,
                                                     const float* __restrict__ x,
                                                     float* __restrict__ y) {
  const int N = 4096, C = 512;
  const int ct = blockIdx.x, ot = blockIdx.y;
  const int batch = ct >> 6, p0 = (ct & 63) * 64, o0 = ot * 32;
  const _Float16* obb = ob + (size_t)batch * C * N;

  v8f acc[2][4];
  {
    v8f z = {};
#pragma unroll
    for (int i = 0; i < 2; ++i)
#pragma unroll
      for (int j = 0; j < 4; ++j) acc[i][j] = z;
  }
  for (int kc = 0; kc < C; kc += 32) {
    v16h a0 = load_frag_a(W + (size_t)(o0 + 0)  * C + kc, C);
    v16h a1 = load_frag_a(W + (size_t)(o0 + 16) * C + kc, C);
#pragma unroll
    for (int j = 0; j < 4; ++j) {
      v16h b = load_frag_b(obb + (size_t)kc * N + p0 + 16 * j, N);
      acc[0][j] = WMMA_F16(a0, b, acc[0][j]);
      acc[1][j] = WMMA_F16(a1, b, acc[1][j]);
    }
  }
  const int lane = threadIdx.x, n = lane & 15, mb = (lane & 16) ? 8 : 0;
#pragma unroll
  for (int i = 0; i < 2; ++i) {
#pragma unroll
    for (int j = 0; j < 4; ++j) {
#pragma unroll
      for (int r = 0; r < 8; ++r) {
        const int o = o0 + 16 * i + mb + r;
        const size_t idx = ((size_t)batch * C + o) * N + p0 + 16 * j + n;
        y[idx] = acc[i][j][r] + bias[o] + x[idx];
      }
    }
  }
}

extern "C" void kernel_launch(void* const* d_in, const int* in_sizes, int n_in,
                              void* d_out, int out_size, void* d_ws, size_t ws_size,
                              hipStream_t stream) {
  (void)in_sizes; (void)n_in; (void)out_size; (void)ws_size;
  const float* x   = (const float*)d_in[0];
  const float* gw  = (const float*)d_in[1];
  const float* gb  = (const float*)d_in[2];
  const float* wq  = (const float*)d_in[3];
  const float* bq  = (const float*)d_in[4];
  const float* wkv = (const float*)d_in[5];
  const float* bkv = (const float*)d_in[6];
  const float* wo  = (const float*)d_in[7];
  const float* bo  = (const float*)d_in[8];
  float* y = (float*)d_out;

  char* ws = (char*)d_ws;
  const size_t SZ = (size_t)2 * 512 * 4096 * sizeof(_Float16);  // 8 MiB each
  _Float16* xn    = (_Float16*)ws; ws += SZ;
  _Float16* q_t   = (_Float16*)ws; ws += SZ;
  _Float16* k_b   = (_Float16*)ws; ws += SZ;
  _Float16* v_t   = (_Float16*)ws; ws += SZ;
  _Float16* o_b   = (_Float16*)ws; ws += SZ;
  _Float16* wq_h  = (_Float16*)ws; ws += (size_t)512 * 512 * sizeof(_Float16);
  _Float16* wkv_h = (_Float16*)ws; ws += (size_t)1024 * 512 * sizeof(_Float16);
  _Float16* wo_h  = (_Float16*)ws;

  f2h_kernel<<<(262144 + 255) / 256, 256, 0, stream>>>(wq, wq_h, 262144);
  f2h_kernel<<<(524288 + 255) / 256, 256, 0, stream>>>(wkv, wkv_h, 524288);
  f2h_kernel<<<(262144 + 255) / 256, 256, 0, stream>>>(wo, wo_h, 262144);

  gn_kernel<<<64, 256, 0, stream>>>(x, gw, gb, xn);

  dim3 gproj(128, 16);
  proj_kernel<<<gproj, 32, 0, stream>>>(wq_h, bq, xn, q_t, 1);                         // q -> [b,h,i,d]
  proj_kernel<<<gproj, 32, 0, stream>>>(wkv_h, bkv, xn, k_b, 0);                       // k -> [b,c,j]
  proj_kernel<<<gproj, 32, 0, stream>>>(wkv_h + (size_t)512 * 512, bkv + 512, xn, v_t, 1); // v -> [b,h,j,d]

  dim3 gattn(128, 16);
  attn_kernel<<<gattn, 32, 0, stream>>>(q_t, k_b, v_t, o_b);

  outproj_kernel<<<gproj, 32, 0, stream>>>(wo_h, bo, o_b, x, y);
}
